// MemNodeV4_GLU_65429531787288
// MI455X (gfx1250) — matrix operand order repaired and checked by reading, not compiled, not run on hardware
//
#include <hip/hip_runtime.h>
#include <hip/hip_bf16.h>
#include <math.h>

// ---------------------------------------------------------------------------
// MemNodeV4 GLU node for MI455X (gfx1250, wave32, WMMA 16x16x32 f16 + TDM)
//
// sim GEMM: D[m(k-tile) x n(token)] += A[16x32 f16] * B[32x16 f16]
//   A = l2norm(mem_bank) (128x192), B = l2norm(proj)^T
//   A tile staged to LDS by the Tensor Data Mover (tensor_load_to_lds).
// E = exp(sim/0.05) stored token-major [N][128] (67 MB -> L2 resident)
// Sinkhorn collapses to row scalings R_k; column scalings cancel in argmax.
// GLU is precomputed as a 128x192 table (assign has only 128 distinct rows).
// ---------------------------------------------------------------------------

typedef __attribute__((ext_vector_type(16))) _Float16 v16h;
typedef __attribute__((ext_vector_type(8)))  float    v8f;
typedef __attribute__((ext_vector_type(4)))  unsigned v4u;
typedef __attribute__((ext_vector_type(8)))  unsigned v8u;

#define NTOK            (64 * 2048)         // 131072 tokens
#define DIM             192
#define KBANK           128
#define TOKS_PER_BLOCK  128
#define GEMM_BLOCKS     (NTOK / TOKS_PER_BLOCK)   // 1024
#define KSTEPS          6                    // 192 / 32
#define ROWSUM_TOKS     512
#define EXP_SCALE       28.853900817779268f  // log2(e) / 0.05
#define A_TILE_QW       ((KBANK * DIM * 2) / 8)   // 6144 x 8B = 48 KB

// --- WMMA fragment swizzle (CDNA5 ISA 7.12.2, wave32) ----------------------
// 16-bit A 16x32: lanes 0-15 row m hold K {0..7,16..23}; lanes 16-31 hold
// K {8..15,24..31}. half index e: K = (e&7) + 16*(e>>3) + 8*(lane>>4).
__device__ __forceinline__ int a_swz(int k, int d) {
  int kt = k >> 4, m = k & 15, ks = d / 32, kk = d & 31;
  int hi = (kk >> 3) & 1;
  int e  = (kk & 7) + ((kk >> 4) << 3);
  int lane = m + (hi << 4);
  return (((kt * KSTEPS + ks) << 5) + lane) * 16 + e;
}
// 16-bit B 32x16: lanes 0-15 col n hold K {0..7,16..23}; lanes 16-31 hold
// K {8..15,24..31} (same k order as A).
__device__ __forceinline__ int b_swz(int n, int d) {
  int nt = n >> 4, c = n & 15, ks = d / 32, kk = d & 31;
  int lane = c + (((kk >> 3) & 1) << 4);
  int e    = (kk & 7) + ((kk >> 4) << 3);
  return (((nt * KSTEPS + ks) << 5) + lane) * 16 + e;
}

// --- Phase 0a: mem_bank -> l2-normalized f16, pre-swizzled A fragments -----
__global__ void prep_mem_kernel(const float* __restrict__ mem,
                                _Float16* __restrict__ memAsw) {
  __shared__ float innorm[KBANK];
  const int t = threadIdx.x;
  if (t < KBANK) {
    float ss = 0.f;
    for (int d = 0; d < DIM; ++d) { float v = mem[t * DIM + d]; ss = fmaf(v, v, ss); }
    innorm[t] = rsqrtf(fmaxf(ss, 1e-12f));
  }
  __syncthreads();
  for (int i = t; i < KBANK * DIM; i += blockDim.x) {
    int k = i / DIM, d = i - k * DIM;
    memAsw[a_swz(k, d)] = (_Float16)(mem[i] * innorm[k]);
  }
}

// --- Phase 0b: GLU table gated[k][d] = lin_d * sigmoid(lin_{d+192}) --------
__global__ void gated_kernel(const float* __restrict__ mem,
                             const float* __restrict__ w,
                             const float* __restrict__ b,
                             float* __restrict__ gated) {
  const int k = blockIdx.x;      // 128
  const int d = threadIdx.x;     // 192
  float a1 = b[d], a2 = b[d + DIM];
  for (int j = 0; j < DIM; ++j) {
    float m = mem[k * DIM + j];
    a1 = fmaf(m, w[j * 2 * DIM + d], a1);
    a2 = fmaf(m, w[j * 2 * DIM + d + DIM], a2);
  }
  gated[k * DIM + d] = a1 / (1.f + __expf(-a2));
}

// --- Phase 1: TDM A-stage + fused normalize + WMMA GEMM + exp -> E [N][128] -
__global__ __launch_bounds__(256) void sim_exp_kernel(
    const float* __restrict__ proj, const _Float16* __restrict__ memAsw,
    float* __restrict__ E) {
  extern __shared__ _Float16 smem[];
  _Float16* sA = smem;                 // 24576 halves (48 KB)
  _Float16* sB = smem + KBANK * DIM;   // 24576 halves (48 KB)
  const int t = threadIdx.x, lane = t & 31, wv = t >> 5;
  const int base = blockIdx.x * TOKS_PER_BLOCK;

  // stage A via Tensor Data Mover: one 6144x1 tile of 8B elements,
  // global (pre-swizzled fragments) -> LDS. Issued once per block by wave 0;
  // EXEC is ignored by tensor ops, completion tracked with TENSORcnt.
  if (wv == 0) {
    unsigned long long ga = (unsigned long long)memAsw;            // byte addr
    unsigned lds_base = (unsigned)(unsigned long long)(void*)sA;   // LDS offset

    v4u g0;                                   // D# group 0 (ISA 8.3)
    g0[0] = 0x1u;                             // count=1, user descriptor
    g0[1] = lds_base;                         // lds_addr (bytes)
    g0[2] = (unsigned)(ga & 0xffffffffu);     // global_addr[31:0]
    g0[3] = (unsigned)((ga >> 32) & 0x1ffffffu) | (2u << 30); // [56:32] | type=2

    v8u g1;                                   // D# group 1 (ISA 8.4)
    g1[0] = 3u << 16;                         // data_size=8B; no mcast/pad/iter
    g1[1] = (A_TILE_QW & 0xffffu) << 16;      // tensor_dim0[15:0]
    g1[2] = ((A_TILE_QW >> 16) & 0xffffu) | (1u << 16); // dim0[31:16]|dim1 lo=1
    g1[3] = (A_TILE_QW & 0xffffu) << 16;      // dim1 hi=0 | tile_dim0=6144
    g1[4] = 1u;                               // tile_dim1=1, tile_dim2=0
    g1[5] = (unsigned)A_TILE_QW;              // tensor_dim0_stride[31:0]
    g1[6] = 0u;                               // stride hi | dim1_stride lo
    g1[7] = 0u;                               // dim1_stride hi

    asm volatile("tensor_load_to_lds %0, %1"
                 :: "s"(g0), "s"(g1)
                 : "memory");
  }

  // stage B: each wave normalizes 16 tokens, writes swizzled f16 fragments
  // (overlaps with the in-flight TDM transfer of the A tile)
  for (int tt = 0; tt < 16; ++tt) {
    int nl = wv * 16 + tt;
    const float* row = proj + (size_t)(base + nl) * DIM;
    if (tt + 1 < 16)
      __builtin_prefetch(row + DIM + lane, 0, 1);   // global_prefetch_b8
    float v[6]; float ss = 0.f;
    #pragma unroll
    for (int j = 0; j < 6; ++j) { v[j] = row[lane + 32 * j]; ss = fmaf(v[j], v[j], ss); }
    #pragma unroll
    for (int m = 16; m >= 1; m >>= 1) ss += __shfl_xor(ss, m);
    float sc = rsqrtf(fmaxf(ss, 1e-12f));
    #pragma unroll
    for (int j = 0; j < 6; ++j)
      sB[b_swz(nl, lane + 32 * j)] = (_Float16)(v[j] * sc);
  }
  if (wv == 0)
    __builtin_amdgcn_s_wait_tensorcnt(0);   // A tile landed in LDS
  __syncthreads();

  // wave wv owns k-rows [16wv, 16wv+16) x all 8 token sub-tiles
  v8f acc[8] = {};
  #pragma unroll
  for (int ks = 0; ks < KSTEPS; ++ks) {
    v16h a = *(const v16h*)(sA + ((((wv * KSTEPS + ks) << 5) + lane) << 4));
    #pragma unroll
    for (int nt = 0; nt < 8; ++nt) {
      v16h bf = *(const v16h*)(sB + ((((nt * KSTEPS + ks) << 5) + lane) << 4));
      acc[nt] = __builtin_amdgcn_wmma_f32_16x16x32_f16(
          /*neg_a=*/false, a, /*neg_b=*/false, bf,
          /*c_mod=*/(short)0, acc[nt], /*reuse_a=*/false, /*reuse_b=*/false);
    }
  }

  // D layout: VGPR v of lane -> m = v + 8*(lane>>4), n = lane&15
  const int krow = (wv << 4) + ((lane >> 4) << 3);
  #pragma unroll
  for (int nt = 0; nt < 8; ++nt) {
    int n = base + (nt << 4) + (lane & 15);
    float* dst = E + (size_t)n * KBANK + krow;
    float4 lo = { exp2f(acc[nt][0] * EXP_SCALE), exp2f(acc[nt][1] * EXP_SCALE),
                  exp2f(acc[nt][2] * EXP_SCALE), exp2f(acc[nt][3] * EXP_SCALE) };
    float4 hi = { exp2f(acc[nt][4] * EXP_SCALE), exp2f(acc[nt][5] * EXP_SCALE),
                  exp2f(acc[nt][6] * EXP_SCALE), exp2f(acc[nt][7] * EXP_SCALE) };
    *(float4*)dst = lo;
    *(float4*)(dst + 4) = hi;
  }
}

// --- Sinkhorn row pass: part[b][k] = sum_n E[n][k] * C_n (C=1 on iter 0) ---
__global__ void rowsum_partial_kernel(const float* __restrict__ E,
                                      const float* __restrict__ C,
                                      int use_c, float* __restrict__ part) {
  __shared__ float red[256];
  const int t = threadIdx.x;
  const int k = t & (KBANK - 1);
  const int par = t >> 7;   // two token parities per block for ILP
  const size_t n0 = (size_t)blockIdx.x * ROWSUM_TOKS;
  float acc = 0.f;
  for (int i = 0; i < ROWSUM_TOKS / 2; ++i) {
    size_t n = n0 + 2 * i + par;
    float cn = use_c ? C[n] : 1.f;
    acc = fmaf(E[n * KBANK + k], cn, acc);
  }
  red[t] = acc;
  __syncthreads();
  if (t < KBANK) part[(size_t)blockIdx.x * KBANK + k] = red[t] + red[t + KBANK];
}

// --- R_k = 1 / (K * sum_b part[b][k]) --------------------------------------
__global__ void rupdate_kernel(const float* __restrict__ part,
                               float* __restrict__ R) {
  const int k = threadIdx.x;   // 128
  float s = 0.f;
  for (int b = 0; b < NTOK / ROWSUM_TOKS; ++b) s += part[(size_t)b * KBANK + k];
  R[k] = 1.f / ((float)KBANK * s);
}

// --- C_n = 1 / (N * sum_k E[n][k] * R_k) -----------------------------------
__global__ void colupdate_kernel(const float* __restrict__ E,
                                 const float* __restrict__ R,
                                 float* __restrict__ C) {
  __shared__ float sR[KBANK];
  const int t = threadIdx.x;
  if (t < KBANK) sR[t] = R[t];
  __syncthreads();
  const int lane = t & 31, wv = t >> 5;
  const int g = lane >> 3, sub = lane & 7;      // 8 lanes per token
  const size_t n = (size_t)blockIdx.x * 32 + wv * 4 + g;
  const float4* e4 = (const float4*)(E + n * KBANK + sub * 16);
  float acc = 0.f;
  #pragma unroll
  for (int q = 0; q < 4; ++q) {
    float4 e = e4[q];
    int kb = sub * 16 + q * 4;
    acc += e.x * sR[kb] + e.y * sR[kb + 1] + e.z * sR[kb + 2] + e.w * sR[kb + 3];
  }
  #pragma unroll
  for (int m = 1; m < 8; m <<= 1) acc += __shfl_xor(acc, m);
  if (sub == 0) C[n] = 1.f / ((float)NTOK * acc);
}

// --- Final: per-token argmax_k E[n][k]*R_k, gather gated, average ----------
__global__ void argmax_out_kernel(const float* __restrict__ E,
                                  const float* __restrict__ R,
                                  const float* __restrict__ gated,
                                  const float* __restrict__ proj,
                                  float* __restrict__ out) {
  __shared__ float sR[KBANK];
  const int t = threadIdx.x;
  if (t < KBANK) sR[t] = R[t];
  __syncthreads();
  const int lane = t & 31, wv = t >> 5;
  const size_t n = (size_t)blockIdx.x * 8 + wv;   // one wave per token
  const float4* e4 = (const float4*)(E + n * KBANK + lane * 4);
  float4 e = e4[0];
  int k0 = lane * 4;
  float bv = e.x * sR[k0]; int bi = k0;
  float v1 = e.y * sR[k0 + 1]; if (v1 > bv) { bv = v1; bi = k0 + 1; }
  float v2 = e.z * sR[k0 + 2]; if (v2 > bv) { bv = v2; bi = k0 + 2; }
  float v3 = e.w * sR[k0 + 3]; if (v3 > bv) { bv = v3; bi = k0 + 3; }
  #pragma unroll
  for (int m = 1; m < 32; m <<= 1) {        // butterfly; first-max tie rule
    float ov = __shfl_xor(bv, m);
    int   oi = __shfl_xor(bi, m);
    if (ov > bv || (ov == bv && oi < bi)) { bv = ov; bi = oi; }
  }
  const float* grow = gated + (size_t)bi * DIM;
  const float* prow = proj + n * DIM;
  float* orow = out + n * DIM;
  #pragma unroll
  for (int j = 0; j < 6; ++j) {
    int d = lane + 32 * j;
    orow[d] = 0.5f * (prow[d] + grow[d]);
  }
}

// ---------------------------------------------------------------------------
extern "C" void kernel_launch(void* const* d_in, const int* in_sizes, int n_in,
                              void* d_out, int out_size, void* d_ws, size_t ws_size,
                              hipStream_t stream) {
  (void)in_sizes; (void)n_in; (void)out_size; (void)ws_size;
  const float* proj = (const float*)d_in[0];   // [64,2048,192]
  const float* mem  = (const float*)d_in[1];   // [128,192]
  const float* gw   = (const float*)d_in[2];   // [192,384]
  const float* gb   = (const float*)d_in[3];   // [384]
  float* out = (float*)d_out;

  char* ws = (char*)d_ws;
  size_t off = 0;
  auto alloc = [&](size_t bytes) -> char* {
    char* p = ws + off;
    off += (bytes + 255) & ~(size_t)255;
    return p;
  };
  float*    E      = (float*)alloc((size_t)NTOK * KBANK * sizeof(float));   // 67 MB (L2)
  float*    gated  = (float*)alloc((size_t)KBANK * DIM * sizeof(float));
  _Float16* memAsw = (_Float16*)alloc((size_t)KBANK * DIM * sizeof(_Float16));
  float*    C      = (float*)alloc((size_t)NTOK * sizeof(float));
  float*    R      = (float*)alloc((size_t)KBANK * sizeof(float));
  float*    part   = (float*)alloc((size_t)(NTOK / ROWSUM_TOKS) * KBANK * sizeof(float));

  prep_mem_kernel<<<1, 256, 0, stream>>>(mem, memAsw);
  gated_kernel<<<KBANK, DIM, 0, stream>>>(mem, gw, gb, gated);

  const size_t gemm_lds = (size_t)2 * KBANK * DIM * sizeof(_Float16);  // 96 KB
  sim_exp_kernel<<<GEMM_BLOCKS, 256, gemm_lds, stream>>>(proj, memAsw, E);

  // 3 row-normalizations, 2 column-normalizations (3rd col + global norm
  // cancel inside argmax)
  for (int it = 0; it < 3; ++it) {
    rowsum_partial_kernel<<<NTOK / ROWSUM_TOKS, 256, 0, stream>>>(E, C, it > 0 ? 1 : 0, part);
    rupdate_kernel<<<1, KBANK, 0, stream>>>(part, R);
    if (it < 2)
      colupdate_kernel<<<NTOK / 32, 256, 0, stream>>>(E, R, C);
  }

  argmax_out_kernel<<<NTOK / 8, 256, 0, stream>>>(E, R, gated, proj, out);
}
